// SGLSP_9216999817293
// MI455X (gfx1250) — matrix operand order repaired and checked by reading, not compiled, not run on hardware
//
#include <hip/hip_runtime.h>
#include <hip/hip_bf16.h>

// ---------------------------------------------------------------------------
// Model dims (fixed by the reference)
// ---------------------------------------------------------------------------
#define B_   64
#define L_   100
#define BL_  (B_ * L_)        // 6400
#define H_   224
#define H3_  672
#define G_   192
#define P_   10000
#define C_   300
#define CPAD_ 304
#define K_   50
#define DU_  64
#define NHEAD_ 4
#define DH_  56

typedef _Float16 v16h __attribute__((ext_vector_type(16)));
typedef _Float16 half8 __attribute__((ext_vector_type(8)));
typedef float    v8f  __attribute__((ext_vector_type(8)));

union F16x16 { v16h v; half8 h[2]; };

// ---------------------------------------------------------------------------
// WMMA GEMM:  C[M,N] = act(A[M,K]f16 @ B[N,K]f16^T + bias (+ res))
// Block = 256 threads = 8 waves arranged 2 (M) x 4 (N).
// Wave tile = 32 M x 32 N = 2x2 accumulators of 16x16 (v_wmma_f32_16x16x32_f16)
// -> per 32-deep K chunk: 8 x global_load_b128 feed 4 WMMAs (A & B frags each
//    reused twice from registers).
// A frag (16x32 f16): lane m = lane%16; elems 0..7  -> K = kc + (lane/16)*8
//                     elems 8..15 -> K = kc + 16 + (lane/16)*8
// B frag (32x16 f16): lane n = lane%16; same K mapping as A frag
// C frag: elem r -> row m0 + r + 8*(lane/16), col n0 + lane%16
// ---------------------------------------------------------------------------
template<int ACT>
__device__ __forceinline__ void wmma_store(
    v8f acc, int m0, int n0, int lane, const float* __restrict__ bias,
    const float* __restrict__ res, float* __restrict__ C, _Float16* __restrict__ C16,
    int M, int Nreal, int ldC)
{
    const int col = n0 + (lane & 15);
    if (col >= Nreal) return;
    const float bv = bias ? bias[col] : 0.0f;
    const int rbase = m0 + (lane >> 4) * 8;
    #pragma unroll
    for (int r = 0; r < 8; ++r) {
        const int row = rbase + r;
        if (row >= M) continue;
        float v = acc[r] + bv;
        if (res) v += res[(size_t)row * ldC + col];
        if (ACT == 1) v = v > 0.0f ? v : 0.0f;
        C[(size_t)row * ldC + col] = v;
        if (C16) C16[(size_t)row * ldC + col] = (_Float16)v;
    }
}

template<int ACT>
__global__ __launch_bounds__(256) void gemm_wmma(
    const _Float16* __restrict__ A, const _Float16* __restrict__ B,
    const float* __restrict__ bias, const float* __restrict__ res,
    float* __restrict__ C, _Float16* __restrict__ C16,
    int M, int Npad, int Nreal, int K, int ldC)
{
    const int wave = threadIdx.x >> 5;
    const int lane = threadIdx.x & 31;

    // wave-uniform tile coordinates (keeps EXEC all-ones through the WMMAs)
    int mt0 = blockIdx.x * 64 + (wave >> 2) * 32;
    int mt1 = mt0 + 16;
    if (mt0 > M - 16) mt0 = M - 16;   // clamp: recompute valid rows (M % 64 != 0)
    if (mt1 > M - 16) mt1 = M - 16;
    const int nt0 = blockIdx.y * 128 + (wave & 3) * 32;
    if (nt0 >= Npad) return;          // wave-uniform early out
    const int  nt1    = nt0 + 16;
    const bool has_n1 = (nt1 < Npad);
    const int  nt1c   = has_n1 ? nt1 : nt0;

    const int hlf = lane >> 4;        // 0 or 1
    const int lr  = lane & 15;
    const _Float16* pa0 = A + (size_t)(mt0 + lr) * K + hlf * 8;
    const _Float16* pa1 = A + (size_t)(mt1 + lr) * K + hlf * 8;
    const _Float16* pb0 = B + (size_t)(nt0 + lr) * K + hlf * 8;
    const _Float16* pb1 = B + (size_t)(nt1c + lr) * K + hlf * 8;

    v8f acc00 = {}, acc01 = {}, acc10 = {}, acc11 = {};
    for (int kc = 0; kc < K; kc += 32) {
        F16x16 a0, a1, b0, b1;
        a0.h[0] = *(const half8*)(pa0 + kc);
        a0.h[1] = *(const half8*)(pa0 + kc + 16);
        a1.h[0] = *(const half8*)(pa1 + kc);
        a1.h[1] = *(const half8*)(pa1 + kc + 16);
        b0.h[0] = *(const half8*)(pb0 + kc);
        b0.h[1] = *(const half8*)(pb0 + kc + 16);
        b1.h[0] = *(const half8*)(pb1 + kc);
        b1.h[1] = *(const half8*)(pb1 + kc + 16);
        __builtin_prefetch(pa0 + kc + 64, 0, 1);   // global_prefetch_b8, 2 chunks ahead
        __builtin_prefetch(pa1 + kc + 64, 0, 1);
        __builtin_prefetch(pb0 + kc + 64, 0, 1);
        __builtin_prefetch(pb1 + kc + 64, 0, 1);
        acc00 = __builtin_amdgcn_wmma_f32_16x16x32_f16(false, a0.v, false, b0.v, (short)0, acc00, false, false);
        acc01 = __builtin_amdgcn_wmma_f32_16x16x32_f16(false, a0.v, false, b1.v, (short)0, acc01, false, false);
        acc10 = __builtin_amdgcn_wmma_f32_16x16x32_f16(false, a1.v, false, b0.v, (short)0, acc10, false, false);
        acc11 = __builtin_amdgcn_wmma_f32_16x16x32_f16(false, a1.v, false, b1.v, (short)0, acc11, false, false);
    }

    wmma_store<ACT>(acc00, mt0, nt0, lane, bias, res, C, C16, M, Nreal, ldC);
    wmma_store<ACT>(acc10, mt1, nt0, lane, bias, res, C, C16, M, Nreal, ldC);
    if (has_n1) {
        wmma_store<ACT>(acc01, mt0, nt1, lane, bias, res, C, C16, M, Nreal, ldC);
        wmma_store<ACT>(acc11, mt1, nt1, lane, bias, res, C, C16, M, Nreal, ldC);
    }
}

// ---------------------------------------------------------------------------
// f32 -> f16 conversion helpers
// ---------------------------------------------------------------------------
__global__ void conv16_kernel(const float* __restrict__ s, _Float16* __restrict__ d, int n) {
    int i = blockIdx.x * blockDim.x + threadIdx.x;
    if (i < n) d[i] = (_Float16)s[i];
}
__global__ void convpad16_kernel(const float* __restrict__ s, _Float16* __restrict__ d,
                                 int rows, int rowspad, int cols) {
    int i = blockIdx.x * blockDim.x + threadIdx.x;
    if (i < rowspad * cols) {
        int r = i / cols;
        d[i] = (r < rows) ? (_Float16)s[i] : (_Float16)0.0f;
    }
}

// ---------------------------------------------------------------------------
// Embedding concat + pad mask + masked x   (one block per (b,l), 224 threads)
// ---------------------------------------------------------------------------
__global__ __launch_bounds__(224) void embed_kernel(
    const int* __restrict__ pois, const int* __restrict__ times, const int* __restrict__ cates,
    const float* __restrict__ ptab, const float* __restrict__ ttab, const float* __restrict__ ctab,
    float* __restrict__ lf, _Float16* __restrict__ lf16,
    float* __restrict__ x, _Float16* __restrict__ x16, int* __restrict__ pad)
{
    const int bl = blockIdx.x, t = threadIdx.x;
    const int poi = pois[bl];
    float v;
    if (t < 128) {
        v = ptab[(size_t)poi * 128 + t];
    } else if (t < 160) {
        const int ti = (times[bl] / 3600) % 168 + 1;
        v = ttab[(size_t)ti * 32 + (t - 128)];
    } else {
        v = ctab[(size_t)cates[bl] * 64 + (t - 160)];
    }
    const size_t o = (size_t)bl * H_ + t;
    lf[o] = v; lf16[o] = (_Float16)v;
    const float xv = (poi == 0) ? 0.0f : v;
    x[o] = xv; x16[o] = (_Float16)xv;
    if (t == 0) pad[bl] = (poi == 0) ? 1 : 0;
}

// ---------------------------------------------------------------------------
// LayerNorm over 224, optional f16 shadow (one block per row, 256 threads)
// ---------------------------------------------------------------------------
__global__ __launch_bounds__(256) void ln_kernel(
    const float* __restrict__ src, const float* __restrict__ g, const float* __restrict__ b,
    float* __restrict__ dst, _Float16* __restrict__ dst16)
{
    const int row = blockIdx.x, t = threadIdx.x;
    __shared__ float red[256];
    float v = (t < H_) ? src[(size_t)row * H_ + t] : 0.0f;
    red[t] = v; __syncthreads();
    for (int o = 128; o > 0; o >>= 1) { if (t < o) red[t] += red[t + o]; __syncthreads(); }
    const float mean = red[0] / (float)H_;
    __syncthreads();
    const float d = (t < H_) ? (v - mean) : 0.0f;
    red[t] = d * d; __syncthreads();
    for (int o = 128; o > 0; o >>= 1) { if (t < o) red[t] += red[t + o]; __syncthreads(); }
    const float rs = rsqrtf(red[0] / (float)H_ + 1e-8f);
    if (t < H_) {
        const float y = d * rs * g[t] + b[t];
        dst[(size_t)row * H_ + t] = y;
        if (dst16) dst16[(size_t)row * H_ + t] = (_Float16)y;
    }
}

// ---------------------------------------------------------------------------
// GRU recurrence: one block per batch element (only the L dimension is serial)
// ---------------------------------------------------------------------------
__global__ __launch_bounds__(256) void gru_kernel(
    const float* __restrict__ xW, const float* __restrict__ Whh,
    const float* __restrict__ bhh, float* __restrict__ sfeat)
{
    const int b = blockIdx.x, t = threadIdx.x;
    __shared__ float h[H_];
    __shared__ float hW[H3_];
    for (int u = t; u < H_; u += 256) h[u] = 0.0f;
    __syncthreads();
    for (int l = 0; l < L_; ++l) {
        const float* xr = xW + (size_t)(b * L_ + l) * H3_;
        for (int j = t; j < H3_; j += 256) {
            const float* w = Whh + (size_t)j * H_;
            float a = bhh[j];
            for (int k = 0; k < H_; ++k) a += h[k] * w[k];
            hW[j] = a;
        }
        __syncthreads();
        for (int u = t; u < H_; u += 256) {
            const float r  = 1.0f / (1.0f + __expf(-(xr[u]        + hW[u])));
            const float z  = 1.0f / (1.0f + __expf(-(xr[H_ + u]   + hW[H_ + u])));
            const float n  = tanhf(xr[2 * H_ + u] + r * hW[2 * H_ + u]);
            const float hn = (1.0f - z) * n + z * h[u];
            h[u] = hn;
            sfeat[(size_t)(b * L_ + l) * H_ + u] = hn;
        }
        __syncthreads();
    }
}

// ---------------------------------------------------------------------------
// Attention: one block per (b, head, query-row); qkv = [Q(224)|K(224)|V(224)]
// ---------------------------------------------------------------------------
__global__ __launch_bounds__(128) void attn_kernel(
    const float* __restrict__ qkv, const int* __restrict__ pad,
    float* __restrict__ ao, _Float16* __restrict__ ao16)
{
    const int idx = blockIdx.x;
    const int l = idx % L_, hh = (idx / L_) % NHEAD_, b = idx / (L_ * NHEAD_);
    __shared__ float s[L_];
    __shared__ float red[128];
    const int t = threadIdx.x;
    const float* qrow = qkv + (size_t)(b * L_ + l) * H3_ + hh * DH_;
    for (int m = t; m < L_; m += 128) {
        float sc;
        if (m > l || pad[b * L_ + m]) sc = -1e9f;
        else {
            const float* krow = qkv + (size_t)(b * L_ + m) * H3_ + H_ + hh * DH_;
            float d = 0.0f;
            for (int g = 0; g < DH_; ++g) d += qrow[g] * krow[g];
            sc = d * 0.13363062095621219f;   // 1/sqrt(56)
        }
        s[m] = sc;
    }
    __syncthreads();
    float mx = -1e30f;
    for (int m = t; m < L_; m += 128) mx = fmaxf(mx, s[m]);
    red[t] = mx; __syncthreads();
    for (int o = 64; o > 0; o >>= 1) { if (t < o) red[t] = fmaxf(red[t], red[t + o]); __syncthreads(); }
    mx = red[0]; __syncthreads();
    float sm = 0.0f;
    for (int m = t; m < L_; m += 128) { const float e = __expf(s[m] - mx); s[m] = e; sm += e; }
    red[t] = sm; __syncthreads();
    for (int o = 64; o > 0; o >>= 1) { if (t < o) red[t] += red[t + o]; __syncthreads(); }
    const float inv = 1.0f / red[0];
    __syncthreads();
    if (t < DH_) {
        float o = 0.0f;
        for (int m = 0; m < L_; ++m)
            o += s[m] * qkv[(size_t)(b * L_ + m) * H3_ + 2 * H_ + hh * DH_ + t];
        o *= inv;
        const size_t oi = (size_t)(b * L_ + l) * H_ + hh * DH_ + t;
        ao[oi] = o; ao16[oi] = (_Float16)o;
    }
}

// ---------------------------------------------------------------------------
// x *= (~pad); refresh f16 shadow
// ---------------------------------------------------------------------------
__global__ __launch_bounds__(224) void mask_kernel(const int* __restrict__ pad,
                                                   float* __restrict__ x, _Float16* __restrict__ x16)
{
    const int bl = blockIdx.x, t = threadIdx.x;
    const size_t o = (size_t)bl * H_ + t;
    const float v = pad[bl] ? 0.0f : x[o];
    x[o] = v; x16[o] = (_Float16)v;
}

// ---------------------------------------------------------------------------
// GAT branch
// ---------------------------------------------------------------------------
__global__ __launch_bounds__(192) void pcf_kernel(
    const float* __restrict__ ptab, const float* __restrict__ ctab, const int* __restrict__ p2c,
    float* __restrict__ pcf, _Float16* __restrict__ pcf16)
{
    const int p = blockIdx.x, t = threadIdx.x;
    float v;
    if (t < 128) v = ptab[(size_t)(p + 1) * 128 + t];
    else         v = ctab[(size_t)p2c[p + 1] * 64 + (t - 128)];
    const size_t o = (size_t)p * G_ + t;
    pcf[o] = v; pcf16[o] = (_Float16)v;
}

__global__ __launch_bounds__(256) void halhar_kernel(
    const float* __restrict__ h, const float* __restrict__ al, const float* __restrict__ ar,
    const float* __restrict__ beta, float* __restrict__ hal, float* __restrict__ har,
    float* __restrict__ bsum)
{
    const int p = blockIdx.x * blockDim.x + threadIdx.x;
    if (p >= P_) return;
    const float* hr = h + (size_t)p * G_;
    float a = 0.0f, r = 0.0f;
    for (int g = 0; g < G_; ++g) { a += hr[g] * al[g]; r += hr[g] * ar[g]; }
    hal[p] = a; har[p] = r;
    float s = 0.0f;
    const float* bp = beta + (size_t)p * K_;
    for (int k = 0; k < K_; ++k) s += bp[k];
    bsum[p] = s;
}

__global__ __launch_bounds__(192) void gat_kernel(
    const float* __restrict__ h, const float* __restrict__ hd,
    const float* __restrict__ hal, const float* __restrict__ har,
    const float* __restrict__ bsum, const int* __restrict__ nbr,
    float* __restrict__ geo)
{
    const int p = blockIdx.x, t = threadIdx.x;
    __shared__ float al[K_];
    __shared__ int   nb[K_];
    __shared__ float mx_s, sm_s;
    if (t < K_) {
        const int idx = nbr[(size_t)p * K_ + t];
        nb[t] = idx;
        const float e = hal[idx] + har[p];
        al[t] = e > 0.0f ? e : 0.2f * e;   // leaky_relu(0.2)
    }
    __syncthreads();
    if (t == 0) { float m = -1e30f; for (int k = 0; k < K_; ++k) m = fmaxf(m, al[k]); mx_s = m; }
    __syncthreads();
    if (t < K_) al[t] = __expf(al[t] - mx_s);
    __syncthreads();
    if (t == 0) { float s = 0.0f; for (int k = 0; k < K_; ++k) s += al[k]; sm_s = s; }
    __syncthreads();
    const float inv = 1.0f / sm_s;
    float acc = h[(size_t)p * G_ + t];
    for (int k = 0; k < K_; ++k) acc += al[k] * inv * h[(size_t)nb[k] * G_ + t];
    const float el = acc > 0.0f ? acc : (__expf(acc) - 1.0f);   // elu
    const float v = (el + bsum[p] * hd[(size_t)p * G_ + t]) * 0.5f;
    geo[(size_t)p * G_ + t] = v > 0.0f ? v : 0.0f;
}

__global__ __launch_bounds__(192) void gtok_kernel(
    const float* __restrict__ geo, const int* __restrict__ pois,
    float* __restrict__ gt, _Float16* __restrict__ gt16)
{
    const int bl = blockIdx.x, t = threadIdx.x;
    const int poi = pois[bl];
    float v = 0.0f;
    if (poi != 0) {
        int r = poi - 1;
        if (r < 0) r = 0;
        if (r > P_ - 1) r = P_ - 1;
        v = geo[(size_t)r * G_ + t];
    }
    gt[(size_t)bl * G_ + t] = v; gt16[(size_t)bl * G_ + t] = (_Float16)v;
}

// ---------------------------------------------------------------------------
// Social branch: per-user projection + LN (one block per batch element)
// ---------------------------------------------------------------------------
__global__ __launch_bounds__(256) void user_kernel(
    const int* __restrict__ users, const float* __restrict__ utab,
    const float* __restrict__ W, const float* __restrict__ bb,
    const float* __restrict__ g, const float* __restrict__ be,
    float* __restrict__ out)
{
    const int b = blockIdx.x, t = threadIdx.x;
    __shared__ float u[DU_];
    __shared__ float red[256];
    if (t < DU_) u[t] = utab[(size_t)users[b] * DU_ + t];
    __syncthreads();
    float v = 0.0f;
    if (t < H_) {
        const float* w = W + (size_t)t * DU_;
        v = bb[t];
        for (int k = 0; k < DU_; ++k) v += u[k] * w[k];
    }
    red[t] = (t < H_) ? v : 0.0f; __syncthreads();
    for (int o = 128; o > 0; o >>= 1) { if (t < o) red[t] += red[t + o]; __syncthreads(); }
    const float mean = red[0] / (float)H_;
    __syncthreads();
    const float d = (t < H_) ? (v - mean) : 0.0f;
    red[t] = d * d; __syncthreads();
    for (int o = 128; o > 0; o >>= 1) { if (t < o) red[t] += red[t + o]; __syncthreads(); }
    const float rs = rsqrtf(red[0] / (float)H_ + 1e-8f);
    if (t < H_) out[(size_t)b * H_ + t] = d * rs * g[t] + be[t];
}

// ---------------------------------------------------------------------------
// final = log_feats + s_feat + geo_feat + user_feat (broadcast over L)
// ---------------------------------------------------------------------------
__global__ __launch_bounds__(224) void final_kernel(
    const float* __restrict__ logf, const float* __restrict__ sf,
    const float* __restrict__ gf, const float* __restrict__ uf,
    float* __restrict__ fin, _Float16* __restrict__ fin16)
{
    const int bl = blockIdx.x, t = threadIdx.x, b = bl / L_;
    const size_t o = (size_t)bl * H_ + t;
    const float v = logf[o] + sf[o] + gf[o] + uf[(size_t)b * H_ + t];
    fin[o] = v; fin16[o] = (_Float16)v;
}

// ---------------------------------------------------------------------------
// Host-side GEMM dispatch
// ---------------------------------------------------------------------------
static void gemm(hipStream_t s, const _Float16* A, const _Float16* B, const float* bias,
                 const float* res, float* C, _Float16* C16,
                 int M, int Npad, int Nreal, int K, int ldC, bool relu)
{
    dim3 grid((M + 63) / 64, (Npad + 127) / 128);
    if (relu) gemm_wmma<1><<<grid, 256, 0, s>>>(A, B, bias, res, C, C16, M, Npad, Nreal, K, ldC);
    else      gemm_wmma<0><<<grid, 256, 0, s>>>(A, B, bias, res, C, C16, M, Npad, Nreal, K, ldC);
}

static void conv16(hipStream_t s, const float* src, _Float16* dst, int n) {
    conv16_kernel<<<(n + 255) / 256, 256, 0, s>>>(src, dst, n);
}

// ---------------------------------------------------------------------------
extern "C" void kernel_launch(void* const* d_in, const int* in_sizes, int n_in,
                              void* d_out, int out_size, void* d_ws, size_t ws_size,
                              hipStream_t stream)
{
    (void)in_sizes; (void)n_in; (void)out_size; (void)ws_size;
    // ---- inputs (setup_inputs order) ----
    const int*   users    = (const int*)  d_in[0];
    const int*   pois     = (const int*)  d_in[1];
    const int*   times    = (const int*)  d_in[2];
    const int*   cates    = (const int*)  d_in[3];
    const int*   nbr_idx  = (const int*)  d_in[4];
    const int*   poi2cate = (const int*)  d_in[5];
    const float* beta     = (const float*)d_in[6];
    const float* user_tab = (const float*)d_in[7];
    const float* poi_tab  = (const float*)d_in[8];
    const float* time_tab = (const float*)d_in[9];
    const float* cate_tab = (const float*)d_in[10];
    const float* gru_Wih  = (const float*)d_in[11];
    const float* gru_Whh  = (const float*)d_in[12];
    const float* gru_bih  = (const float*)d_in[13];
    const float* gru_bhh  = (const float*)d_in[14];
    const float* ln1_g    = (const float*)d_in[15];
    const float* ln1_b    = (const float*)d_in[16];
    const float* Wqkv     = (const float*)d_in[17];
    const float* bqkv     = (const float*)d_in[18];
    const float* Wo       = (const float*)d_in[19];
    const float* bo       = (const float*)d_in[20];
    const float* ln2_g    = (const float*)d_in[21];
    const float* ln2_b    = (const float*)d_in[22];
    const float* ff_W1    = (const float*)d_in[23];
    const float* ff_b1    = (const float*)d_in[24];
    const float* ff_W2    = (const float*)d_in[25];
    const float* ff_b2    = (const float*)d_in[26];
    const float* last_g   = (const float*)d_in[27];
    const float* last_b   = (const float*)d_in[28];
    const float* gat_W    = (const float*)d_in[29];
    const float* gat_al   = (const float*)d_in[30];
    const float* gat_ar   = (const float*)d_in[31];
    const float* w_dist   = (const float*)d_in[32];
    const float* geo_W    = (const float*)d_in[33];
    const float* geo_b    = (const float*)d_in[34];
    const float* geo_lng  = (const float*)d_in[35];
    const float* geo_lnb  = (const float*)d_in[36];
    const float* user_W   = (const float*)d_in[37];
    const float* user_b   = (const float*)d_in[38];
    const float* user_lng = (const float*)d_in[39];
    const float* user_lnb = (const float*)d_in[40];
    const float* out_W    = (const float*)d_in[41];
    const float* out_b    = (const float*)d_in[42];
    const float* outc_W   = (const float*)d_in[43];
    const float* outc_b   = (const float*)d_in[44];

    float* out_main = (float*)d_out;                       // [B,L,P]
    float* out_cat  = (float*)d_out + (size_t)BL_ * P_;    // [B,L,C]

    // ---- workspace bump allocator ----
    char* base = (char*)d_ws;
    size_t off = 0;
    auto alloc = [&](size_t bytes) -> char* {
        char* p = base + off;
        off += (bytes + 255) & ~(size_t)255;
        return p;
    };
    const size_t BLH = (size_t)BL_ * H_;
    const size_t BL3H = (size_t)BL_ * H3_;
    const size_t PG  = (size_t)P_ * G_;
    const size_t BLG = (size_t)BL_ * G_;

    float* lf      = (float*)alloc(BLH * 4);
    float* x       = (float*)alloc(BLH * 4);
    float* Q       = (float*)alloc(BLH * 4);
    float* Y       = (float*)alloc(BLH * 4);
    float* T1      = (float*)alloc(BLH * 4);
    float* ao      = (float*)alloc(BLH * 4);
    float* sfeat   = (float*)alloc(BLH * 4);
    float* logf    = (float*)alloc(BLH * 4);
    float* geo_pre = (float*)alloc(BLH * 4);
    float* geo_ft  = (float*)alloc(BLH * 4);
    float* fin     = (float*)alloc(BLH * 4);
    float* xWqkv   = (float*)alloc(BL3H * 4);      // xW for GRU, then reused as qkv
    float* pcf     = (float*)alloc(PG * 4);
    float* hbuf    = (float*)alloc(PG * 4);
    float* hdbuf   = (float*)alloc(PG * 4);
    float* geo_all = (float*)alloc(PG * 4);
    float* gtok    = (float*)alloc(BLG * 4);
    float* ufeat   = (float*)alloc((size_t)B_ * H_ * 4);
    float* hal     = (float*)alloc((size_t)P_ * 4);
    float* har     = (float*)alloc((size_t)P_ * 4);
    float* bsum    = (float*)alloc((size_t)P_ * 4);
    int*   pad     = (int*)  alloc((size_t)BL_ * 4);

    _Float16* lf16   = (_Float16*)alloc(BLH * 2);
    _Float16* x16    = (_Float16*)alloc(BLH * 2);
    _Float16* Q16    = (_Float16*)alloc(BLH * 2);
    _Float16* Y16    = (_Float16*)alloc(BLH * 2);
    _Float16* T116   = (_Float16*)alloc(BLH * 2);
    _Float16* ao16   = (_Float16*)alloc(BLH * 2);
    _Float16* fin16  = (_Float16*)alloc(BLH * 2);
    _Float16* gt16   = (_Float16*)alloc(BLG * 2);
    _Float16* pcf16  = (_Float16*)alloc(PG * 2);
    _Float16* Wih16  = (_Float16*)alloc((size_t)H3_ * H_ * 2);
    _Float16* Wqkv16 = (_Float16*)alloc((size_t)2 * H3_ * H_ * 2);
    _Float16* Wo16   = (_Float16*)alloc((size_t)2 * H_ * H_ * 2);
    _Float16* W116   = (_Float16*)alloc((size_t)2 * H_ * H_ * 2);
    _Float16* W216   = (_Float16*)alloc((size_t)2 * H_ * H_ * 2);
    _Float16* gatW16 = (_Float16*)alloc((size_t)G_ * G_ * 2);
    _Float16* wdst16 = (_Float16*)alloc((size_t)G_ * G_ * 2);
    _Float16* geoW16 = (_Float16*)alloc((size_t)H_ * G_ * 2);
    _Float16* outW16 = (_Float16*)alloc((size_t)P_ * H_ * 2);
    _Float16* outc16 = (_Float16*)alloc((size_t)CPAD_ * H_ * 2);

    // ---- weight conversions to f16 ----
    conv16(stream, gru_Wih, Wih16, H3_ * H_);
    conv16(stream, Wqkv,    Wqkv16, 2 * H3_ * H_);
    conv16(stream, Wo,      Wo16,   2 * H_ * H_);
    conv16(stream, ff_W1,   W116,   2 * H_ * H_);
    conv16(stream, ff_W2,   W216,   2 * H_ * H_);
    conv16(stream, gat_W,   gatW16, G_ * G_);
    conv16(stream, w_dist,  wdst16, G_ * G_);
    conv16(stream, geo_W,   geoW16, H_ * G_);
    conv16(stream, out_W,   outW16, P_ * H_);
    convpad16_kernel<<<(CPAD_ * H_ + 255) / 256, 256, 0, stream>>>(outc_W, outc16, C_, CPAD_, H_);

    // ---- embeddings / masks ----
    embed_kernel<<<BL_, 224, 0, stream>>>(pois, times, cates, poi_tab, time_tab, cate_tab,
                                          lf, lf16, x, x16, pad);

    // ---- GRU: xW GEMM then batch-parallel recurrence ----
    gemm(stream, lf16, Wih16, gru_bih, nullptr, xWqkv, nullptr, BL_, H3_, H3_, H_, H3_, false);
    gru_kernel<<<B_, 256, 0, stream>>>(xWqkv, gru_Whh, gru_bhh, sfeat);

    // ---- GAT / geo branch ----
    pcf_kernel<<<P_, 192, 0, stream>>>(poi_tab, cate_tab, poi2cate, pcf, pcf16);
    gemm(stream, pcf16, gatW16, nullptr, nullptr, hbuf,  nullptr, P_, G_, G_, G_, G_, false);
    gemm(stream, pcf16, wdst16, nullptr, nullptr, hdbuf, nullptr, P_, G_, G_, G_, G_, false);
    halhar_kernel<<<(P_ + 255) / 256, 256, 0, stream>>>(hbuf, gat_al, gat_ar, beta, hal, har, bsum);
    gat_kernel<<<P_, 192, 0, stream>>>(hbuf, hdbuf, hal, har, bsum, nbr_idx, geo_all);
    gtok_kernel<<<BL_, 192, 0, stream>>>(geo_all, pois, gtok, gt16);
    gemm(stream, gt16, geoW16, geo_b, nullptr, geo_pre, nullptr, BL_, H_, H_, G_, H_, false);
    ln_kernel<<<BL_, 256, 0, stream>>>(geo_pre, geo_lng, geo_lnb, geo_ft, nullptr);

    // ---- social branch ----
    user_kernel<<<B_, 256, 0, stream>>>(users, user_tab, user_W, user_b, user_lng, user_lnb, ufeat);

    // ---- transformer (2 layers); xWqkv reused as qkv buffer (GRU already done) ----
    float* qkv = xWqkv;
    for (int i = 0; i < 2; ++i) {
        ln_kernel<<<BL_, 256, 0, stream>>>(x, ln1_g + i * H_, ln1_b + i * H_, Q, Q16);
        // Q projection from LN'd q; K,V projections from raw x
        gemm(stream, Q16, Wqkv16 + (size_t)i * H3_ * H_, bqkv + i * H3_, nullptr,
             qkv, nullptr, BL_, H_, H_, H_, H3_, false);
        gemm(stream, x16, Wqkv16 + (size_t)i * H3_ * H_ + (size_t)H_ * H_, bqkv + i * H3_ + H_,
             nullptr, qkv + H_, nullptr, BL_, 2 * H_, 2 * H_, H_, H3_, false);
        attn_kernel<<<B_ * NHEAD_ * L_, 128, 0, stream>>>(qkv, pad, ao, ao16);
        gemm(stream, ao16, Wo16 + (size_t)i * H_ * H_, bo + i * H_, Q,
             x, nullptr, BL_, H_, H_, H_, H_, false);                      // x = Q + proj(o)
        ln_kernel<<<BL_, 256, 0, stream>>>(x, ln2_g + i * H_, ln2_b + i * H_, Y, Y16);
        gemm(stream, Y16, W116 + (size_t)i * H_ * H_, ff_b1 + i * H_, nullptr,
             T1, T116, BL_, H_, H_, H_, H_, true);                          // relu
        gemm(stream, T116, W216 + (size_t)i * H_ * H_, ff_b2 + i * H_, Y,
             x, nullptr, BL_, H_, H_, H_, H_, false);                       // x = Y + ffn
        mask_kernel<<<BL_, 224, 0, stream>>>(pad, x, x16);
    }
    ln_kernel<<<BL_, 256, 0, stream>>>(x, last_g, last_b, logf, nullptr);

    // ---- fuse branches, project to outputs ----
    final_kernel<<<BL_, 224, 0, stream>>>(logf, sfeat, geo_ft, ufeat, fin, fin16);
    gemm(stream, fin16, outW16, out_b,  nullptr, out_main, nullptr, BL_, P_,    P_, H_, P_, false);
    gemm(stream, fin16, outc16, outc_b, nullptr, out_cat,  nullptr, BL_, CPAD_, C_, H_, C_, false);
}
